// MultiHeadSelfAttention_28913719836744
// MI455X (gfx1250) — compile-verified, matrix operand order checked by reading, not run on hardware
//
#include <hip/hip_runtime.h>
#include <hip/hip_bf16.h>

#define D_MODEL   1024
#define NUM_HEADS 16
#define HEAD_DIM  64
#define SEQ       2048
#define BATCH     2
#define MTOT      (BATCH * SEQ) /* 4096 */

typedef __attribute__((ext_vector_type(16))) __bf16 v16bf;
typedef __attribute__((ext_vector_type(8)))  float  v8f;

// ---------- small helpers ----------

__device__ __forceinline__ unsigned short f2bf_bits(float f) {
  union { float f; unsigned u; } a; a.f = f;
  unsigned r = (a.u + 0x7fffu + ((a.u >> 16) & 1u)) >> 16; // round-to-nearest-even
  return (unsigned short)r;
}
__device__ __forceinline__ __bf16 f2bf(float f) {
  union { unsigned short s; __bf16 b; } o; o.s = f2bf_bits(f);
  return o.b;
}

__device__ __forceinline__ v8f wmma_bf16(v16bf a, v16bf b, v8f c) {
  // v_wmma_f32_16x16x32_bf16  D = A*B + C  (f32 accum)
  return __builtin_amdgcn_wmma_f32_16x16x32_bf16(false, a, false, b, (short)0, c,
                                                 false, false);
}

// A/B fragment (16x32 bf16, K-contiguous rows in memory).
// ISA layout: lane&15 = row(M)/col(N); elems 0..7 = K k0..k0+7,
// elems 8..15 = K k0+16..k0+23, with k0 = (lane<16 ? 0 : 8).
__device__ __forceinline__ v16bf frag_global(const __bf16* __restrict__ row, int k, int lane) {
  const int k0 = k + ((lane & 16) ? 8 : 0);
  union { uint4 u[2]; v16bf v; } t;
  t.u[0] = *(const uint4*)(row + k0);
  t.u[1] = *(const uint4*)(row + k0 + 16);
  return t.v;
}

// Same fragment pattern out of LDS (8B-granular, padded row strides).
__device__ __forceinline__ v16bf frag_lds(const __bf16* base, int strideElems,
                                          int row, int kcol, int lane) {
  const __bf16* r = base + row * strideElems + kcol + ((lane & 16) ? 8 : 0);
  const uint2* p = (const uint2*)r;
  union { uint2 u[4]; v16bf v; } t;
  t.u[0] = p[0]; t.u[1] = p[1];   // K k0 .. k0+7
  t.u[2] = p[4]; t.u[3] = p[5];   // K k0+16 .. k0+23
  return t.v;
}

// ---------- kernel 1: f32 -> bf16 convert ----------

__global__ __launch_bounds__(256) void cvt_bf16(const float* __restrict__ in,
                                                __bf16* __restrict__ out, int n) {
  int i = (blockIdx.x * 256 + threadIdx.x) * 4;
  if (i + 3 < n) {
    float4 f = *(const float4*)(in + i);
    union { unsigned short s[4]; uint2 u; } pk;
    pk.s[0] = f2bf_bits(f.x); pk.s[1] = f2bf_bits(f.y);
    pk.s[2] = f2bf_bits(f.z); pk.s[3] = f2bf_bits(f.w);
    *(uint2*)(out + i) = pk.u;
  }
}

// ---------- shared 128x128 GEMM core: C = A(MxK) * W(NxK)^T ----------
// Block: 256 threads = 8 waves, waves tiled (wm,wn) = (wave>>1, wave&1);
// each wave owns a 32x64 tile = 2x4 grid of 16x16 WMMA accumulators.

__device__ __forceinline__ void gemm_core_128x128(const __bf16* __restrict__ A,
                                                  const __bf16* __restrict__ W,
                                                  int Mb, int Nb, int lane,
                                                  int wm, int wn, v8f acc[2][4]) {
  const __bf16* a0 = A + (size_t)(Mb + wm * 32 + (lane & 15)) * D_MODEL;
  const __bf16* a1 = a0 + 16 * D_MODEL;
  const __bf16* b0 = W + (size_t)(Nb + wn * 64 + (lane & 15)) * D_MODEL;
  const __bf16* b1 = b0 + 16 * D_MODEL;
  const __bf16* b2 = b0 + 32 * D_MODEL;
  const __bf16* b3 = b0 + 48 * D_MODEL;
  for (int k = 0; k < D_MODEL; k += 32) {
    __builtin_prefetch(a0 + k + 256, 0, 1);
    __builtin_prefetch(b0 + k + 256, 0, 1);
    v16bf fa0 = frag_global(a0, k, lane);
    v16bf fa1 = frag_global(a1, k, lane);
    v16bf fb0 = frag_global(b0, k, lane);
    v16bf fb1 = frag_global(b1, k, lane);
    v16bf fb2 = frag_global(b2, k, lane);
    v16bf fb3 = frag_global(b3, k, lane);
    acc[0][0] = wmma_bf16(fa0, fb0, acc[0][0]);
    acc[0][1] = wmma_bf16(fa0, fb1, acc[0][1]);
    acc[0][2] = wmma_bf16(fa0, fb2, acc[0][2]);
    acc[0][3] = wmma_bf16(fa0, fb3, acc[0][3]);
    acc[1][0] = wmma_bf16(fa1, fb0, acc[1][0]);
    acc[1][1] = wmma_bf16(fa1, fb1, acc[1][1]);
    acc[1][2] = wmma_bf16(fa1, fb2, acc[1][2]);
    acc[1][3] = wmma_bf16(fa1, fb3, acc[1][3]);
  }
}

// ---------- kernel 2: fused QKV projection + bias + RoPE ----------
// blockIdx.z selects Q/K/V. Writes bf16 into head-major [B*H, S, 64].

__global__ __launch_bounds__(256) void qkv_rope_gemm(
    const __bf16* __restrict__ xb,
    const __bf16* __restrict__ Wqb, const __bf16* __restrict__ Wkb,
    const __bf16* __restrict__ Wvb,
    const float* __restrict__ bq, const float* __restrict__ bk,
    const float* __restrict__ bv,
    __bf16* __restrict__ Qo, __bf16* __restrict__ Ko, __bf16* __restrict__ Vo) {
  const int mode = blockIdx.z;
  const __bf16* W    = (mode == 0) ? Wqb : (mode == 1) ? Wkb : Wvb;
  const float*  bias = (mode == 0) ? bq  : (mode == 1) ? bk  : bv;
  __bf16*       out  = (mode == 0) ? Qo  : (mode == 1) ? Ko  : Vo;

  const int lane = threadIdx.x & 31;
  const int wave = threadIdx.x >> 5;
  const int wm = wave >> 1, wn = wave & 1;
  const int Mb = blockIdx.x * 128, Nb = blockIdx.y * 128;

  v8f acc[2][4];
#pragma unroll
  for (int i = 0; i < 2; ++i)
#pragma unroll
    for (int j = 0; j < 4; ++j) acc[i][j] = (v8f){0.f,0.f,0.f,0.f,0.f,0.f,0.f,0.f};

  gemm_core_128x128(xb, W, Mb, Nb, lane, wm, wn, acc);

  // Epilogue: bias (+ RoPE for Q/K). C layout: N = lane&15; VGPR v holds
  // rows M=v (lanes 0-15) and M=v+8 (lanes 16-31).
#pragma unroll
  for (int mt = 0; mt < 2; ++mt) {
#pragma unroll
    for (int nt = 0; nt < 4; ++nt) {
      const int n = Nb + wn * 64 + nt * 16 + (lane & 15);
      const int h = n >> 6, d = n & (HEAD_DIM - 1);
      const float bsv = bias[n];
      // inv_freq = theta^{-(d&~1)/hd}
      const float inv_freq =
          __expf((float)(d & ~1) * (-9.2103403719761836f / (float)HEAD_DIM));
#pragma unroll
      for (int v = 0; v < 8; ++v) {
        const int m = Mb + wm * 32 + mt * 16 + v + ((lane & 16) ? 8 : 0);
        const int bb = m >> 11;            // / SEQ
        const int s  = m & (SEQ - 1);
        float val = acc[mt][nt][v] + bsv;
        if (mode < 2) {                    // RoPE on Q and K
          const float ang = (float)s * inv_freq;
          const float cs = __cosf(ang), sn = __sinf(ang);
          const float other = __shfl_xor(val, 1, 32); // pair partner (d ^ 1)
          val = (lane & 1) ? (other * sn + val * cs)  // odd d: t1*sin + t2*cos
                           : (val * cs - other * sn); // even d: t1*cos - t2*sin
        }
        out[(((size_t)(bb * NUM_HEADS + h)) * SEQ + s) * HEAD_DIM + d] = f2bf(val);
      }
    }
  }
}

// ---------- kernel 3: causal flash attention ----------
// Block = 128 queries (8 waves x 16 rows), loop over 32-key tiles.
// K tile and V^T tile staged in LDS with padded strides; P goes through
// per-wave LDS scratch to convert C-layout -> A-layout.

__global__ __launch_bounds__(256) void flash_attn(
    const __bf16* __restrict__ Q, const __bf16* __restrict__ K,
    const __bf16* __restrict__ V, __bf16* __restrict__ O) {
  __shared__ __bf16 sK[32 * 72];       // 32 keys x 64 d   (+8 pad)
  __shared__ __bf16 sVT[64 * 36];      // 64 d    x 32 keys(+4 pad), transposed
  __shared__ __bf16 sP[8 * 16 * 36];   // per-wave 16 q x 32 keys (+4 pad)

  const int lane = threadIdx.x & 31;
  const int wave = threadIdx.x >> 5;
  const int bh    = blockIdx.y;        // b*H + h
  const int qbase = blockIdx.x * 128;
  const int qrow0 = qbase + wave * 16;

  const __bf16* Qh = Q + (size_t)bh * SEQ * HEAD_DIM;
  const __bf16* Kh = K + (size_t)bh * SEQ * HEAD_DIM;
  const __bf16* Vh = V + (size_t)bh * SEQ * HEAD_DIM;

  const __bf16* qrow = Qh + (size_t)(qrow0 + (lane & 15)) * HEAD_DIM;
  const v16bf aq0 = frag_global(qrow, 0, lane);   // d = 0..31
  const v16bf aq1 = frag_global(qrow, 32, lane);  // d = 32..63

  v8f o0 = (v8f){0.f,0.f,0.f,0.f,0.f,0.f,0.f,0.f};
  v8f o1 = o0, o2 = o0, o3 = o0;
  float mi[8], li[8];
#pragma unroll
  for (int v = 0; v < 8; ++v) { mi[v] = -1e30f; li[v] = 0.0f; }

  __bf16* pw = &sP[wave * 16 * 36];
  const int kvend = qbase + 128;

  for (int kv = 0; kv < kvend; kv += 32) {
    __syncthreads();  // protect sK/sVT from previous iteration's readers
    {   // stage K tile: 32 rows x 64 bf16, row stride 72 (conflict-free)
      const int j = threadIdx.x >> 3, c = threadIdx.x & 7;
      union { uint4 a; uint2 b[2]; } u;
      u.a = *(const uint4*)(Kh + (size_t)(kv + j) * HEAD_DIM + c * 8);
      uint2* dst = (uint2*)(sK + j * 72 + c * 8);
      dst[0] = u.b[0]; dst[1] = u.b[1];
    }
    {   // stage V tile transposed: sVT[d][j], row stride 36
      const int j = threadIdx.x & 31, db = threadIdx.x >> 5;
      union { uint4 a; __bf16 h[8]; } u;
      u.a = *(const uint4*)(Vh + (size_t)(kv + j) * HEAD_DIM + db * 8);
#pragma unroll
      for (int e = 0; e < 8; ++e) sVT[(db * 8 + e) * 36 + j] = u.h[e];
    }
    __syncthreads();

    const bool active = (kv <= qrow0 + 15);  // wave-uniform
    if (active) {
      // ---- scores: S = Q * K^T (16q x 32 keys), 4 WMMAs ----
      v8f s0 = (v8f){0.f,0.f,0.f,0.f,0.f,0.f,0.f,0.f};
      v8f s1 = s0;
      {
        v16bf bk0 = frag_lds(sK, 72,       (lane & 15), 0, lane);
        v16bf bk1 = frag_lds(sK, 72,       (lane & 15), 32, lane);
        s0 = wmma_bf16(aq0, bk0, s0);
        s0 = wmma_bf16(aq1, bk1, s0);
        v16bf bk2 = frag_lds(sK, 72, 16 + (lane & 15), 0, lane);
        v16bf bk3 = frag_lds(sK, 72, 16 + (lane & 15), 32, lane);
        s1 = wmma_bf16(aq0, bk2, s1);
        s1 = wmma_bf16(aq1, bk3, s1);
      }
      // ---- online softmax per row (rows v / v+8 match C layout halves) ----
#pragma unroll
      for (int v = 0; v < 8; ++v) {
        const int q  = qrow0 + v + ((lane & 16) ? 8 : 0);
        const int j0 = kv + (lane & 15);
        const float x0 = (j0      <= q) ? s0[v] * 0.125f : -1e30f;
        const float x1 = (j0 + 16 <= q) ? s1[v] * 0.125f : -1e30f;
        float mx = fmaxf(mi[v], fmaxf(x0, x1));
        mx = fmaxf(mx, __shfl_xor(mx, 1, 32));
        mx = fmaxf(mx, __shfl_xor(mx, 2, 32));
        mx = fmaxf(mx, __shfl_xor(mx, 4, 32));
        mx = fmaxf(mx, __shfl_xor(mx, 8, 32));
        const float alpha = __expf(mi[v] - mx);
        const float p0 = __expf(x0 - mx);
        const float p1 = __expf(x1 - mx);
        float rs = p0 + p1;
        rs += __shfl_xor(rs, 1, 32);
        rs += __shfl_xor(rs, 2, 32);
        rs += __shfl_xor(rs, 4, 32);
        rs += __shfl_xor(rs, 8, 32);
        li[v] = li[v] * alpha + rs;
        mi[v] = mx;
        o0[v] *= alpha; o1[v] *= alpha; o2[v] *= alpha; o3[v] *= alpha;
        const int prow = v + ((lane & 16) ? 8 : 0);
        pw[prow * 36      + (lane & 15)] = f2bf(p0);
        pw[prow * 36 + 16 + (lane & 15)] = f2bf(p1);
      }
    }
    __syncthreads();  // publish sP (uniform control flow)
    if (active) {
      // ---- O += P * V : A = P (16x32) from LDS, B = V^T rows from LDS ----
      const v16bf pa = frag_lds(pw, 36, (lane & 15), 0, lane);
      v16bf bv0 = frag_lds(sVT, 36,       (lane & 15), 0, lane);
      v16bf bv1 = frag_lds(sVT, 36, 16 + (lane & 15), 0, lane);
      v16bf bv2 = frag_lds(sVT, 36, 32 + (lane & 15), 0, lane);
      v16bf bv3 = frag_lds(sVT, 36, 48 + (lane & 15), 0, lane);
      o0 = wmma_bf16(pa, bv0, o0);
      o1 = wmma_bf16(pa, bv1, o1);
      o2 = wmma_bf16(pa, bv2, o2);
      o3 = wmma_bf16(pa, bv3, o3);
    }
  }

  // ---- finalize: O /= l, write bf16 into [B, S, D] for the O-projection ----
  const int b = bh >> 4, h = bh & (NUM_HEADS - 1);
#pragma unroll
  for (int v = 0; v < 8; ++v) {
    const float inv = 1.0f / li[v];
    const int q = qrow0 + v + ((lane & 16) ? 8 : 0);
    __bf16* orow = O + ((size_t)(b * SEQ + q)) * D_MODEL + h * HEAD_DIM + (lane & 15);
    orow[0]  = f2bf(o0[v] * inv);
    orow[16] = f2bf(o1[v] * inv);
    orow[32] = f2bf(o2[v] * inv);
    orow[48] = f2bf(o3[v] * inv);
  }
}

// ---------- kernel 4: output projection, fp32 out ----------

__global__ __launch_bounds__(256) void oproj_gemm(const __bf16* __restrict__ Ab,
                                                  const __bf16* __restrict__ Wob,
                                                  const float* __restrict__ bo,
                                                  float* __restrict__ out) {
  const int lane = threadIdx.x & 31;
  const int wave = threadIdx.x >> 5;
  const int wm = wave >> 1, wn = wave & 1;
  const int Mb = blockIdx.x * 128, Nb = blockIdx.y * 128;

  v8f acc[2][4];
#pragma unroll
  for (int i = 0; i < 2; ++i)
#pragma unroll
    for (int j = 0; j < 4; ++j) acc[i][j] = (v8f){0.f,0.f,0.f,0.f,0.f,0.f,0.f,0.f};

  gemm_core_128x128(Ab, Wob, Mb, Nb, lane, wm, wn, acc);

#pragma unroll
  for (int mt = 0; mt < 2; ++mt) {
#pragma unroll
    for (int nt = 0; nt < 4; ++nt) {
      const int n = Nb + wn * 64 + nt * 16 + (lane & 15);
      const float bsv = bo[n];
#pragma unroll
      for (int v = 0; v < 8; ++v) {
        const int m = Mb + wm * 32 + mt * 16 + v + ((lane & 16) ? 8 : 0);
        out[(size_t)m * D_MODEL + n] = acc[mt][nt][v] + bsv;
      }
    }
  }
}

// ---------- host launcher ----------

extern "C" void kernel_launch(void* const* d_in, const int* in_sizes, int n_in,
                              void* d_out, int out_size, void* d_ws, size_t ws_size,
                              hipStream_t stream) {
  const float* x  = (const float*)d_in[0];
  const float* Wq = (const float*)d_in[1];
  const float* bq = (const float*)d_in[2];
  const float* Wk = (const float*)d_in[3];
  const float* bk = (const float*)d_in[4];
  const float* Wv = (const float*)d_in[5];
  const float* bv = (const float*)d_in[6];
  const float* Wo = (const float*)d_in[7];
  const float* bo = (const float*)d_in[8];

  const size_t NX = (size_t)MTOT * D_MODEL;     // 4,194,304
  const size_t NW = (size_t)D_MODEL * D_MODEL;  // 1,048,576

  __bf16* p   = (__bf16*)d_ws;   // 48 MiB total
  __bf16* xb  = p; p += NX;
  __bf16* Wqb = p; p += NW;
  __bf16* Wkb = p; p += NW;
  __bf16* Wvb = p; p += NW;
  __bf16* Wob = p; p += NW;
  __bf16* Qb  = p; p += NX;
  __bf16* Kb  = p; p += NX;
  __bf16* Vb  = p; p += NX;
  __bf16* Ob  = p; p += NX;

  cvt_bf16<<<(int)(NX / 4 / 256), 256, 0, stream>>>(x,  xb,  (int)NX);
  cvt_bf16<<<(int)(NW / 4 / 256), 256, 0, stream>>>(Wq, Wqb, (int)NW);
  cvt_bf16<<<(int)(NW / 4 / 256), 256, 0, stream>>>(Wk, Wkb, (int)NW);
  cvt_bf16<<<(int)(NW / 4 / 256), 256, 0, stream>>>(Wv, Wvb, (int)NW);
  cvt_bf16<<<(int)(NW / 4 / 256), 256, 0, stream>>>(Wo, Wob, (int)NW);

  qkv_rope_gemm<<<dim3(MTOT / 128, D_MODEL / 128, 3), 256, 0, stream>>>(
      xb, Wqb, Wkb, Wvb, bq, bk, bv, Qb, Kb, Vb);

  flash_attn<<<dim3(SEQ / 128, BATCH * NUM_HEADS), 256, 0, stream>>>(Qb, Kb, Vb, Ob);

  oproj_gemm<<<dim3(MTOT / 128, D_MODEL / 128), 256, 0, stream>>>(
      Ob, Wob, bo, (float*)d_out);
}
